// MultiHeadAttention_4509715661567
// MI455X (gfx1250) — matrix operand order repaired and checked by reading, not compiled, not run on hardware
//
#include <hip/hip_runtime.h>
#include <hip/hip_bf16.h>

// ---------------------------------------------------------------------------
// MHA forward for MI455X (gfx1250, wave32, WMMA + Tensor Data Mover).
// Pipeline:  f32 -> bf16 casts  ->  3x WMMA projection GEMMs (bf16, f32 acc)
//            -> fused flash-attention with TDM-staged K/V tiles in LDS.
// ---------------------------------------------------------------------------

typedef __bf16 bf16;
typedef __attribute__((ext_vector_type(16))) __bf16 v16bf;
typedef __attribute__((ext_vector_type(8)))  float  v8f;
typedef __attribute__((ext_vector_type(8)))  unsigned int v8u;
typedef __attribute__((ext_vector_type(4)))  unsigned int u32x4;
typedef __attribute__((ext_vector_type(8)))  int i32x8;
typedef __attribute__((ext_vector_type(4)))  int i32x4;

union FragAB { v8u u; v16bf b; };

#define NUMH   16
#define DK     64
#define HID    1024
#define BATCH  8
#define SEQ    1024
#define MROWS  (BATCH * SEQ)          // 8192

#if defined(__has_builtin)
#  if __has_builtin(__builtin_amdgcn_tensor_load_to_lds)
#    define HAVE_TDM 1
#  endif
#endif
#ifndef HAVE_TDM
#  define HAVE_TDM 0
#endif

// Even K index held by VGPR slot s (0..7) for a 16-bit A-fragment (contiguous: k = 16*half + 2*s)
__device__ __forceinline__ int a_k0(int s, int half) {
    return half * 16 + 2 * s;
}

// 16 contiguous bf16 -> B-fragment (requires K-contiguous storage per column)
__device__ __forceinline__ FragAB load16(const bf16* p) {
    FragAB f;
    uint4 t0 = *(const uint4*)(p);
    uint4 t1 = *(const uint4*)(p + 8);
    f.u[0] = t0.x; f.u[1] = t0.y; f.u[2] = t0.z; f.u[3] = t0.w;
    f.u[4] = t1.x; f.u[5] = t1.y; f.u[6] = t1.z; f.u[7] = t1.w;
    return f;
}

__device__ __forceinline__ v8f wmma_bf16(const FragAB& a, const FragAB& b, v8f c) {
    return __builtin_amdgcn_wmma_f32_16x16x32_bf16(false, a.b, false, b.b,
                                                   (short)0, c, false, false);
}

#if HAVE_TDM
// Tensor Data Mover: 2-D tile load, bf16 elements (data_size code 1 = 2 bytes).
// D# built per CDNA5 ISA.
// This toolchain exposes the 6-arg builtin:
//   (uint32x4 g0, int32x8 g1, int32x4 g2, int32x4 g3, int32x8 extra, i32 cpol)
__device__ __forceinline__ void tdm_load_2d(unsigned int lds_addr,
                                            const void* gptr,
                                            unsigned int tensor_d0,
                                            unsigned int tensor_d1,
                                            unsigned int stride0,
                                            unsigned int tile0,
                                            unsigned int tile1) {
    unsigned long long ga = (unsigned long long)gptr;
    u32x4 g0;
    g0[0] = 1u;                                  // count=1, user descriptor
    g0[1] = lds_addr;                            // LDS byte address
    g0[2] = (unsigned int)(ga & 0xFFFFFFFFu);    // global_addr[31:0]
    g0[3] = (unsigned int)(ga >> 32) | (2u << 30); // global_addr[56:32], type=2
    i32x8 g1;
    g1[0] = (int)(1u << 16);                     // wg_mask=0, data_size=1 (2B)
    g1[1] = (int)((tensor_d0 & 0xFFFFu) << 16);  // tensor_dim0[15:0]
    g1[2] = (int)((tensor_d0 >> 16) | ((tensor_d1 & 0xFFFFu) << 16));
    g1[3] = (int)((tensor_d1 >> 16) | (tile0 << 16)); // tile_dim0
    g1[4] = (int)(tile1 & 0xFFFFu);              // tile_dim1, tile_dim2=0
    g1[5] = (int)stride0;                        // tensor_dim0_stride lo32
    g1[6] = 0;                                   // stride hi, dim1_stride lo
    g1[7] = 0;
    i32x4 z4 = {0, 0, 0, 0};
    i32x8 z8 = {0, 0, 0, 0, 0, 0, 0, 0};
    __builtin_amdgcn_tensor_load_to_lds(g0, g1, z4, z4, z8, 0);
}
#endif

// Stage one 32-key tile of K ([32][64] bf16) and V^T ([64][32] bf16) into LDS.
// Kg points at K-head row kb (tile contiguous); Vg at Vt-head column kb.
__device__ __forceinline__ void stage_kv(const bf16* __restrict__ Kg,
                                         const bf16* __restrict__ Vg,
                                         bf16* kb_lds, bf16* vb_lds,
                                         int tid, int wave) {
#if HAVE_TDM
    if (wave == 0) {
        tdm_load_2d((unsigned int)(unsigned long long)kb_lds, Kg,
                    DK, SEQ, DK, DK, 32);        // 32 rows x 64 cols
        tdm_load_2d((unsigned int)(unsigned long long)vb_lds, Vg,
                    SEQ, DK, SEQ, 32, DK);       // 64 rows x 32 cols
    }
#else
    // cooperative fallback: 2x b128 per tensor per thread (128 threads)
    #pragma unroll
    for (int r = 0; r < 2; ++r) {
        int li = tid + r * 128;                  // 0..255
        ((uint4*)kb_lds)[li] = ((const uint4*)Kg)[li];
        int row = li >> 2, part = li & 3;        // V rows of 32 bf16 = 4x b128
        ((uint4*)vb_lds)[li] = *((const uint4*)(Vg + row * SEQ) + part);
    }
#endif
}

__device__ __forceinline__ void stage_wait(int wave) {
#if HAVE_TDM
    if (wave == 0) __builtin_amdgcn_s_wait_tensorcnt(0);
#endif
}

// ------------------------------ cast kernels -------------------------------

__global__ void cast_f32_bf16(const float* __restrict__ src,
                              bf16* __restrict__ dst, int n) {
    int i = (blockIdx.x * blockDim.x + threadIdx.x) * 4;
    if (i + 3 < n) {
        float4 v = *(const float4*)(src + i);
        dst[i + 0] = (bf16)v.x; dst[i + 1] = (bf16)v.y;
        dst[i + 2] = (bf16)v.z; dst[i + 3] = (bf16)v.w;
    }
}

// Wt[n][k] = (bf16) W[k][n]   (1024x1024), tiled through LDS
__global__ void transpose_cast(const float* __restrict__ W,
                               bf16* __restrict__ Wt) {
    __shared__ float tile[32][33];
    int bx = blockIdx.x * 32, by = blockIdx.y * 32;
    int tx = threadIdx.x, ty = threadIdx.y;               // block (32,8)
    #pragma unroll
    for (int r = ty; r < 32; r += 8)
        tile[r][tx] = W[(by + r) * HID + bx + tx];
    __syncthreads();
    #pragma unroll
    for (int r = ty; r < 32; r += 8)
        Wt[(bx + r) * HID + by + tx] = (bf16)tile[tx][r];
}

// --------------------------- projection GEMM -------------------------------
// Out[M=8192, N=1024] = X[M,1024] @ W[1024,1024] + bias,  W given transposed.
// One wave handles a 16x64 output strip (1 A-frag reused over 4 B-frags).
// vstore=1 scatters the result head-transposed for V: [b][h][d][s2].

__global__ void gemm_proj(const bf16* __restrict__ X,
                          const bf16* __restrict__ Wt,   // [N][K]
                          const float* __restrict__ bias,
                          bf16* __restrict__ Out, int vstore) {
    const int lane = threadIdx.x & 31;
    const int wave = threadIdx.x >> 5;
    const int task = blockIdx.x * 4 + wave;    // 512 * 16 = 8192 tasks
    const int tm   = task >> 4;                // M-tile   [0,512)
    const int tg   = task & 15;                // N-group  [0,16) of 4 tiles
    const int half = lane >> 4;
    const int l16  = lane & 15;
    const int mrow = tm * 16 + l16;
    const int kb2  = half * 16;

    v8f acc[4] = {v8f{}, v8f{}, v8f{}, v8f{}};

    for (int kb = 0; kb < HID; kb += 32) {
        FragAB a;
        #pragma unroll
        for (int s = 0; s < 8; ++s)
            a.u[s] = *(const unsigned int*)(X + mrow * HID + kb + a_k0(s, half));
        #pragma unroll
        for (int g = 0; g < 4; ++g) {
            int ncol = (tg * 4 + g) * 16 + l16;
            FragAB bfr = load16(Wt + ncol * HID + kb + kb2);
            acc[g] = wmma_bf16(a, bfr, acc[g]);
        }
    }

    #pragma unroll
    for (int g = 0; g < 4; ++g) {
        int ncol = (tg * 4 + g) * 16 + l16;
        float bv = bias[ncol];
        #pragma unroll
        for (int i = 0; i < 8; ++i) {
            int mg = tm * 16 + 8 * half + i;           // global row = b*SEQ+s
            float v = acc[g][i] + bv;
            if (!vstore) {
                Out[mg * HID + ncol] = (bf16)v;
            } else {
                int b  = mg >> 10, s = mg & 1023;
                int h  = s >> 6;                        // reshape head split
                int s2 = (s & 63) * 16 + (ncol >> 6);
                int d  = ncol & 63;
                Out[(((b * NUMH + h) << 6) + d) * SEQ + s2] = (bf16)v;
            }
        }
    }
}

// ---------------------------- flash attention ------------------------------
// One block (4 waves) per (b, h, 64-row q-strip); wave w owns one 16-row
// q-tile. K/V 32-key tiles are DMA'd once per block into LDS by the TDM,
// double-buffered so the DMA for tile i+1 overlaps WMMA work on tile i.
// Q/K heads: contiguous [1024][64] bf16; V head-transposed [64][1024] bf16.

__global__ void flash_attn(const bf16* __restrict__ Q,
                           const bf16* __restrict__ K,
                           const bf16* __restrict__ Vt,
                           float* __restrict__ Out) {
    __shared__ __align__(16) bf16 kbuf[2][32 * DK];    // [key][d]   4KB each
    __shared__ __align__(16) bf16 vbuf[2][DK * 32];    // [d][key]   4KB each
    __shared__ __align__(16) bf16 p_lds[4][16 * 32];   // per-wave P scratch

    const int tid  = threadIdx.x;
    const int lane = tid & 31;
    const int wave = tid >> 5;
    const int bh   = blockIdx.x >> 4;                  // 0..127
    const int qg   = blockIdx.x & 15;                  // q-strip in head
    const int qt   = qg * 4 + wave;                    // this wave's q-tile
    const int h    = bh & 15;
    const int b    = bh >> 4;
    const long hb  = (long)bh << 16;                   // head base (elements)
    const int half = lane >> 4;
    const int l16  = lane & 15;
    const int kb2  = half * 16;
    const int qrow = qt * 16 + l16;

    // Q fragments for d-chunks [0,32) and [32,64) — reused for all key tiles
    FragAB qf[2];
    #pragma unroll
    for (int c = 0; c < 2; ++c)
        #pragma unroll
        for (int s = 0; s < 8; ++s)
            qf[c].u[s] = *(const unsigned int*)(Q + hb + qrow * DK + c * 32 +
                                                a_k0(s, half));

    v8f of[4] = {v8f{}, v8f{}, v8f{}, v8f{}};
    float mrow[8], lrow[8];
    #pragma unroll
    for (int i = 0; i < 8; ++i) { mrow[i] = -INFINITY; lrow[i] = 0.f; }

    const float scale = 0.125f;                        // 1/sqrt(64)

    // prologue: stage key-tile 0 into buffer 0
    stage_kv(K + hb, Vt + hb, kbuf[0], vbuf[0], tid, wave);
    stage_wait(wave);
    __syncthreads();

    for (int it = 0; it < SEQ / 32; ++it) {
        const int kb  = it * 32;
        const int cur = it & 1;
        const int nxt = cur ^ 1;

        // prefetch next tile into the other buffer (overlaps WMMA below)
        if (it + 1 < SEQ / 32)
            stage_kv(K + hb + (long)(kb + 32) * DK, Vt + hb + (kb + 32),
                     kbuf[nxt], vbuf[nxt], tid, wave);

        const bf16* kc = kbuf[cur];
        const bf16* vc = vbuf[cur];

        // ---- scores: two 16x16 tiles (keys kb..+15, kb+16..+31) ----
        v8f sc0 = {}, sc1 = {};
        #pragma unroll
        for (int c = 0; c < 2; ++c) {
            FragAB kf0 = load16(kc + l16 * DK        + c * 32 + kb2);
            FragAB kf1 = load16(kc + (16 + l16) * DK + c * 32 + kb2);
            sc0 = wmma_bf16(qf[c], kf0, sc0);
            sc1 = wmma_bf16(qf[c], kf1, sc1);
        }

        // ---- online softmax (rows live across 16 lanes of a half) ----
        #pragma unroll
        for (int i = 0; i < 8; ++i) {
            float s0 = sc0[i] * scale;
            float s1 = sc1[i] * scale;
            float tm = fmaxf(s0, s1);
            #pragma unroll
            for (int m = 8; m >= 1; m >>= 1)
                tm = fmaxf(tm, __shfl_xor(tm, m, 32));
            float mnew  = fmaxf(mrow[i], tm);
            float alpha = __expf(mrow[i] - mnew);
            float p0    = __expf(s0 - mnew);
            float p1    = __expf(s1 - mnew);
            float rs    = p0 + p1;
            #pragma unroll
            for (int m = 8; m >= 1; m >>= 1)
                rs += __shfl_xor(rs, m, 32);
            lrow[i] = lrow[i] * alpha + rs;
            mrow[i] = mnew;
            int r = 8 * half + i;                      // C-layout row
            p_lds[wave][r * 32 + l16]      = (bf16)p0;
            p_lds[wave][r * 32 + 16 + l16] = (bf16)p1;
            of[0][i] *= alpha; of[1][i] *= alpha;
            of[2][i] *= alpha; of[3][i] *= alpha;
        }

        // ---- P: C-layout -> A-layout via per-wave LDS round trip ----
        FragAB pf;
        #pragma unroll
        for (int s = 0; s < 8; ++s)
            pf.u[s] = *(const unsigned int*)&p_lds[wave][l16 * 32 + a_k0(s, half)];

        // ---- O += P (16x32) @ V_tile (32x64) ----
        #pragma unroll
        for (int f = 0; f < 4; ++f) {
            FragAB vf = load16(vc + (16 * f + l16) * 32 + kb2);
            of[f] = wmma_bf16(pf, vf, of[f]);
        }

        // DMA for next tile must land before anyone reads it; all waves must
        // be done with buffers before the following iteration's prefetch.
        stage_wait(wave);
        __syncthreads();
    }

    // ---- normalize and store f32 output: out[b][s2][h*64+d] ----
    #pragma unroll
    for (int f = 0; f < 4; ++f)
        #pragma unroll
        for (int i = 0; i < 8; ++i) {
            int s2 = qt * 16 + 8 * half + i;
            int d  = 16 * f + l16;
            Out[((long)b * SEQ + s2) * HID + h * DK + d] = of[f][i] / lrow[i];
        }
}

// ------------------------------- launcher ----------------------------------

extern "C" void kernel_launch(void* const* d_in, const int* in_sizes, int n_in,
                              void* d_out, int out_size, void* d_ws, size_t ws_size,
                              hipStream_t stream) {
    const float* q  = (const float*)d_in[0];
    const float* k  = (const float*)d_in[1];
    const float* v  = (const float*)d_in[2];
    const float* Wq = (const float*)d_in[3];
    const float* bq = (const float*)d_in[4];
    const float* Wk = (const float*)d_in[5];
    const float* bk = (const float*)d_in[6];
    const float* Wv = (const float*)d_in[7];
    const float* bv = (const float*)d_in[8];
    float* out = (float*)d_out;

    const size_t NACT = (size_t)MROWS * HID;   // 8,388,608 elements
    const size_t NW   = (size_t)HID * HID;     // 1,048,576 elements
    // ws layout (bf16): 3 cast activations + 3 transposed weights + Q,K,Vt
    // total = (3+3)*NACT*2 + 3*NW*2  ~ 107 MB
    bf16* ws  = (bf16*)d_ws;
    bf16* Xq  = ws;
    bf16* Xk  = Xq  + NACT;
    bf16* Xv  = Xk  + NACT;
    bf16* Wqt = Xv  + NACT;
    bf16* Wkt = Wqt + NW;
    bf16* Wvt = Wkt + NW;
    bf16* Qp  = Wvt + NW;
    bf16* Kp  = Qp  + NACT;
    bf16* Vt  = Kp  + NACT;

    // 1) cast activations f32 -> bf16 (4 elems/thread)
    cast_f32_bf16<<<NACT / 4 / 256, 256, 0, stream>>>(q, Xq, (int)NACT);
    cast_f32_bf16<<<NACT / 4 / 256, 256, 0, stream>>>(k, Xk, (int)NACT);
    cast_f32_bf16<<<NACT / 4 / 256, 256, 0, stream>>>(v, Xv, (int)NACT);

    // 2) transpose+cast weights so B-fragments are K-contiguous (b128 loads)
    dim3 tb(32, 8), tg(32, 32);
    transpose_cast<<<tg, tb, 0, stream>>>(Wq, Wqt);
    transpose_cast<<<tg, tb, 0, stream>>>(Wk, Wkt);
    transpose_cast<<<tg, tb, 0, stream>>>(Wv, Wvt);

    // 3) WMMA projections (8192 wave-tasks each; 128-thread blocks = 4 waves)
    gemm_proj<<<2048, 128, 0, stream>>>(Xq, Wqt, bq, Qp, 0);
    gemm_proj<<<2048, 128, 0, stream>>>(Xk, Wkt, bk, Kp, 0);
    gemm_proj<<<2048, 128, 0, stream>>>(Xv, Wvt, bv, Vt, 1);  // head-transposed

    // 4) fused flash attention with TDM-staged K/V, f32 output
    flash_attn<<<2048, 128, 0, stream>>>(Qp, Kp, Vt, out);
}
